// AttentionHead_46377056862698
// MI455X (gfx1250) — matrix operand order repaired and checked by reading, not compiled, not run on hardware
//
#include <hip/hip_runtime.h>
#include <hip/hip_bf16.h>

// ---------------------------------------------------------------------------
// AttentionHead decoder for MI455X (gfx1250, wave32, WMMA).
// bf16 data path for all GEMMs (v_wmma_f32_16x16x32_bf16, f32 accumulate);
// fp32 for softmax/LSTM state math. batch_H + batch_h_proj kept in bf16 so
// the 26-step attention loop's working set (134 MB) fits in the 192 MB L2.
// GEMM inner loop uses GLOBAL_LOAD_ASYNC_TO_LDS_B128 with double-buffered
// LDS tiles and s_wait_asynccnt-based pipelining (CDNA5 async-copy path).
// ---------------------------------------------------------------------------

typedef __hip_bfloat16 bf16_t;
typedef __attribute__((ext_vector_type(16))) __bf16 v16bf;
typedef __attribute__((ext_vector_type(8)))  float  v8f;

#define BDIM   512
#define SDIM   128
#define DDIM   512
#define HDIM   512
#define CCLS   38
#define TSTEPS 26
#define G4H    2048          // 4*H
#define XPAD   576           // context(512) + onehot(38) padded to 576 (mult of 32)
#define KCAT   1088          // XPAD + HDIM

// GEMM tiling: block = 256 threads = 8 waves (2 M-waves x 4 N-waves),
// wave tile 32x32 (2x2 WMMA 16x16 accumulators), block tile 64x128, K step 32.
#define BM 64
#define BN 128
#define BK 32
#define LDS_STRIDE 40        // 32 halves + 8 pad (80 B rows, 16B aligned chunks)

__device__ inline void storev(float* p, float v)  { *p = v; }
__device__ inline void storev(bf16_t* p, float v) { *p = __float2bfloat16(v); }

// tanh via a single hardware v_exp_f32: tanh(x) = sign(x)*(1-e)/(1+e), e=exp(-2|x|)
__device__ __forceinline__ float fast_tanh(float x) {
    const float e = __expf(-2.0f * fabsf(x));
    return copysignf((1.0f - e) / (1.0f + e), x);
}
__device__ __forceinline__ float fast_sigmoid(float x) {
    return 1.0f / (1.0f + __expf(-x));
}

// Async global->LDS 128-bit copy (ASYNCcnt-tracked DMA, no VGPR staging).
// VDST = per-lane LDS byte address, VADDR = per-lane 64-bit global address.
__device__ __forceinline__ void async_b128(const bf16_t* gsrc, bf16_t* ldst) {
    asm volatile("global_load_async_to_lds_b128 %0, %1, off"
                 :: "v"((unsigned)(size_t)ldst), "v"(gsrc)
                 : "memory");
}
template <int N>
__device__ __forceinline__ void wait_asynccnt() {
#if defined(__has_builtin) && __has_builtin(__builtin_amdgcn_s_wait_asynccnt)
    __builtin_amdgcn_s_wait_asynccnt(N);
#else
    asm volatile("s_wait_asynccnt %0" :: "i"(N) : "memory");
#endif
}

// ---------------------------------------------------------------------------
// Generic WMMA GEMM:  Cout[M,N] = A[M,K](bf16,row-major,lda) @ W[N,K]^T + bias
// W stored [N,K] row-major (weight layout), i.e. lane-n fragment = W row n.
// M % 64 == 0, K % 32 == 0 at all call sites. N edge handled by clamping the
// W source row (columns >= N are computed but never stored: nlim guard).
// ---------------------------------------------------------------------------
template <typename OutT>
__global__ void __launch_bounds__(256)
gemm_bf16_wmma(const bf16_t* __restrict__ A, int lda,
               const bf16_t* __restrict__ W,
               const float* __restrict__ bias,
               OutT* __restrict__ Cout, int ldc, int nlim,
               int M, int N, int K)
{
    __shared__ bf16_t sA[2][BM * LDS_STRIDE];
    __shared__ bf16_t sB[2][BN * LDS_STRIDE];

    const int tid  = threadIdx.x;
    const int lane = tid & 31;
    const int wave = tid >> 5;
    const int wm   = wave >> 2;          // 0..1  (M waves)
    const int wn   = wave & 3;           // 0..3  (N waves)
    const int block_m = blockIdx.y * BM;
    const int block_n = blockIdx.x * BN;

    // per-thread staging coordinates (uniform across the wave -> uniform
    // async-issue counts: 1 A-chunk + 2 W-chunks = 3 per thread per tile)
    const int a_row = tid >> 2;                     // 0..63
    const int a_chk = (tid & 3) * 8;                // half offset 0,8,16,24
    const int b_row = tid >> 1;                     // 0..127
    const int b_chk = (tid & 1) * 16;               // half offset 0,16
    const int b_gn  = block_n + b_row;
    const int b_gnc = (b_gn < N) ? b_gn : (N - 1);  // clamp: cols >= N unstored
    const bf16_t* aSrc = A + (size_t)(block_m + a_row) * lda + a_chk;
    const bf16_t* wSrc = W + (size_t)b_gnc * K + b_chk;

    v8f acc[2][2] = {};

    // A fragment K pattern (ISA 7.12.2, 16-bit A 16x32):
    //   lanes 0-15: K = {0..7, 16..23}; lanes 16-31: K = {8..15, 24..31}
    const int a_kbase = (lane & 16) ? 8 : 0;
    // B fragment (dense 32x16): lane n = lane&15; same K pattern as A:
    //   lanes 0-15: K = {0..7, 16..23}; lanes 16-31: K = {8..15, 24..31}
    const int b_kbase = (lane & 16) ? 8 : 0;
    const int fr      = lane & 15;       // M row (A) / N col (B)

    const int ntiles = K / BK;

    // prologue: stage tile 0 into buffer 0
    async_b128(aSrc, &sA[0][a_row * LDS_STRIDE + a_chk]);
    async_b128(wSrc,     &sB[0][b_row * LDS_STRIDE + b_chk]);
    async_b128(wSrc + 8, &sB[0][b_row * LDS_STRIDE + b_chk + 8]);

    for (int i = 0; i < ntiles; ++i) {
        const int cur = i & 1;
        if (i + 1 < ntiles) {
            const int k1 = (i + 1) * BK;
            async_b128(aSrc + k1, &sA[cur ^ 1][a_row * LDS_STRIDE + a_chk]);
            async_b128(wSrc + k1,     &sB[cur ^ 1][b_row * LDS_STRIDE + b_chk]);
            async_b128(wSrc + k1 + 8, &sB[cur ^ 1][b_row * LDS_STRIDE + b_chk + 8]);
            if (i + 2 < ntiles)
                __builtin_prefetch(aSrc + (i + 2) * BK, 0, 3);
            wait_asynccnt<3>();   // tile i landed (async loads complete in order)
        } else {
            wait_asynccnt<0>();
        }
        __syncthreads();          // all waves' tile-i data visible in LDS

        #pragma unroll
        for (int mi = 0; mi < 2; ++mi) {
            const int arow = wm * 32 + mi * 16 + fr;
            union { uint4 u[2]; v16bf v; } af;
            af.u[0] = *(const uint4*)(&sA[cur][arow * LDS_STRIDE + a_kbase]);
            af.u[1] = *(const uint4*)(&sA[cur][arow * LDS_STRIDE + a_kbase + 16]);
            #pragma unroll
            for (int ni = 0; ni < 2; ++ni) {
                const int brow = wn * 32 + ni * 16 + fr;
                union { uint4 u[2]; v16bf v; } bfm;
                bfm.u[0] = *(const uint4*)(&sB[cur][brow * LDS_STRIDE + b_kbase]);
                bfm.u[1] = *(const uint4*)(&sB[cur][brow * LDS_STRIDE + b_kbase + 16]);
                acc[mi][ni] = __builtin_amdgcn_wmma_f32_16x16x32_bf16(
                    false, af.v, false, bfm.v, (short)0, acc[mi][ni], false, false);
            }
        }
        __syncthreads();          // protect buf[cur] before it is re-filled
    }

    // store: C/D layout — VGPR r: lanes 0-15 -> (M=r, N=lane), 16-31 -> (M=8+r)
    const int rbase = (lane & 16) ? 8 : 0;
    const int ncol  = lane & 15;
    #pragma unroll
    for (int mi = 0; mi < 2; ++mi) {
        #pragma unroll
        for (int ni = 0; ni < 2; ++ni) {
            const int gn = block_n + wn * 32 + ni * 16 + ncol;
            if (gn >= nlim) continue;
            const float bv = bias ? bias[gn] : 0.0f;
            #pragma unroll
            for (int r = 0; r < 8; ++r) {
                const int gm = block_m + wm * 32 + mi * 16 + rbase + r;
                storev(&Cout[(size_t)gm * ldc + gn], acc[mi][ni][r] + bv);
            }
        }
    }
}

// ---------------------------------------------------------------------------
// prep kernels
// ---------------------------------------------------------------------------
__global__ void __launch_bounds__(256)
f32_to_bf16(const float* __restrict__ in, bf16_t* __restrict__ out, size_t n)
{
    size_t i = (size_t)blockIdx.x * 256 + threadIdx.x;
    if (i < n) out[i] = __float2bfloat16(in[i]);
}

__global__ void __launch_bounds__(256)
build_wcat(const float* __restrict__ w_ih, const float* __restrict__ w_hh,
           bf16_t* __restrict__ wcat)
{
    int idx = blockIdx.x * 256 + threadIdx.x;             // G4H*KCAT
    if (idx >= G4H * KCAT) return;
    const int g = idx / KCAT, k = idx % KCAT;
    float v = 0.0f;
    if (k < DDIM + CCLS)      v = w_ih[(size_t)g * (DDIM + CCLS) + k];
    else if (k >= XPAD)       v = w_hh[(size_t)g * HDIM + (k - XPAD)];
    wcat[idx] = __float2bfloat16(v);
}

__global__ void __launch_bounds__(256)
build_small(const float* __restrict__ w_gen, const float* __restrict__ b_gen,
            const float* __restrict__ b_ih, const float* __restrict__ b_hh,
            bf16_t* __restrict__ wgen_pad, float* __restrict__ bgen_pad,
            float* __restrict__ bias_cat)
{
    int idx = blockIdx.x * 256 + threadIdx.x;
    if (idx < 48 * HDIM) {
        const int n = idx / HDIM, k = idx % HDIM;
        wgen_pad[idx] = __float2bfloat16(n < CCLS ? w_gen[n * HDIM + k] : 0.0f);
    }
    if (idx < G4H) bias_cat[idx] = b_ih[idx] + b_hh[idx];
    if (idx < 48)  bgen_pad[idx] = (idx < CCLS) ? b_gen[idx] : 0.0f;
}

__global__ void __launch_bounds__(256)
init_state(bf16_t* __restrict__ xh, float* __restrict__ c)
{
    int idx = blockIdx.x * 256 + threadIdx.x;
    if (idx < BDIM * KCAT) xh[idx] = __float2bfloat16(0.0f);
    if (idx < BDIM * HDIM) c[idx] = 0.0f;
}

// ---------------------------------------------------------------------------
// attention: e[b,s] = sum_h tanh(bhp[b,s,h] + hproj[b,h]) * w_score[h]
// then softmax over s. One block per b (8 waves, 16 rows/wave).
// ---------------------------------------------------------------------------
__global__ void __launch_bounds__(256)
attn_softmax_kernel(const bf16_t* __restrict__ bhp,
                    const bf16_t* __restrict__ hproj,
                    const float* __restrict__ w_score,
                    float* __restrict__ alpha)
{
    const int b   = blockIdx.x;
    const int tid = threadIdx.x;
    __shared__ float s_h[HDIM];
    __shared__ float s_ws[HDIM];
    __shared__ float s_e[SDIM];
    __shared__ float red[2];

    for (int i = tid; i < HDIM; i += 256) {
        s_h[i]  = __bfloat162float(hproj[(size_t)b * HDIM + i]);
        s_ws[i] = w_score[i];
    }
    __syncthreads();

    const int wave = tid >> 5, lane = tid & 31;
    const bf16_t* base = bhp + (size_t)b * SDIM * HDIM;
    for (int s = wave; s < SDIM; s += 8) {
        const bf16_t* row = base + (size_t)s * HDIM;
        float acc = 0.0f;
        for (int h = lane; h < HDIM; h += 32)
            acc += fast_tanh(__bfloat162float(row[h]) + s_h[h]) * s_ws[h];
        for (int off = 16; off > 0; off >>= 1)
            acc += __shfl_xor(acc, off, 32);
        if (lane == 0) s_e[s] = acc;
    }
    __syncthreads();

    if (tid < 32) {
        float m = -3.402823466e38f;
        for (int s = lane; s < SDIM; s += 32) m = fmaxf(m, s_e[s]);
        for (int off = 16; off > 0; off >>= 1) m = fmaxf(m, __shfl_xor(m, off, 32));
        float sum = 0.0f;
        for (int s = lane; s < SDIM; s += 32) sum += __expf(s_e[s] - m);
        for (int off = 16; off > 0; off >>= 1) sum += __shfl_xor(sum, off, 32);
        if (lane == 0) { red[0] = m; red[1] = sum; }
    }
    __syncthreads();

    const float m = red[0], inv = 1.0f / red[1];
    for (int s = tid; s < SDIM; s += 256)
        alpha[(size_t)b * SDIM + s] = __expf(s_e[s] - m) * inv;
}

// ---------------------------------------------------------------------------
// context[b,d] = sum_s alpha[b,s]*batch_H[b,s,d]  -> xh[b, 0..511] (bf16)
// plus one-hot(gt[b,t]) -> xh[b, 512..549], zeros -> 550..575
// ---------------------------------------------------------------------------
__global__ void __launch_bounds__(256)
context_onehot_kernel(const float* __restrict__ alpha,
                      const bf16_t* __restrict__ bH,
                      const int* __restrict__ gt, int t,
                      bf16_t* __restrict__ xh)
{
    const int b = blockIdx.x, tid = threadIdx.x;
    __shared__ float s_a[SDIM];
    if (tid < SDIM) s_a[tid] = alpha[(size_t)b * SDIM + tid];
    __syncthreads();

    const bf16_t* base = bH + (size_t)b * SDIM * DDIM;
    for (int d = tid; d < DDIM; d += 256) {
        float acc = 0.0f;
        #pragma unroll 4
        for (int s = 0; s < SDIM; ++s)
            acc += s_a[s] * __bfloat162float(base[(size_t)s * DDIM + d]);
        xh[(size_t)b * KCAT + d] = __float2bfloat16(acc);
    }
    const int label = gt[b * TSTEPS + t];
    for (int j = tid; j < XPAD - DDIM; j += 256)
        xh[(size_t)b * KCAT + DDIM + j] =
            __float2bfloat16((j < CCLS && j == label) ? 1.0f : 0.0f);
}

// ---------------------------------------------------------------------------
// LSTM pointwise: gates [B,4H] (i|f|g|o), update c, emit h (bf16) into the
// activation row (for next step's GEMMs) and into the hiddens buffer.
// ---------------------------------------------------------------------------
__global__ void __launch_bounds__(256)
lstm_update_kernel(const float* __restrict__ gates,
                   float* __restrict__ c,
                   bf16_t* __restrict__ xh,
                   bf16_t* __restrict__ hiddens, int t)
{
    const int idx = blockIdx.x * 256 + threadIdx.x;     // B*H
    const int b = idx >> 9, j = idx & 511;
    const float* g = gates + (size_t)b * G4H;
    const float ig = g[j], fg = g[HDIM + j], gg = g[2 * HDIM + j], og = g[3 * HDIM + j];
    const float si = fast_sigmoid(ig);
    const float sf = fast_sigmoid(fg);
    const float so = fast_sigmoid(og);
    const float cn = sf * c[idx] + si * fast_tanh(gg);
    const float hn = so * fast_tanh(cn);
    c[idx] = cn;
    const bf16_t hb = __float2bfloat16(hn);
    xh[(size_t)b * KCAT + XPAD + j] = hb;
    hiddens[((size_t)b * TSTEPS + t) * HDIM + j] = hb;
}

// ---------------------------------------------------------------------------
// launch
// ---------------------------------------------------------------------------
extern "C" void kernel_launch(void* const* d_in, const int* in_sizes, int n_in,
                              void* d_out, int out_size, void* d_ws, size_t ws_size,
                              hipStream_t stream)
{
    const float* batch_H = (const float*)d_in[0];
    const int*   gt      = (const int*)  d_in[1];
    const float* w_i2h   = (const float*)d_in[2];
    const float* w_h2h   = (const float*)d_in[3];
    const float* b_h2h   = (const float*)d_in[4];
    const float* w_score = (const float*)d_in[5];
    const float* w_ih    = (const float*)d_in[6];
    const float* w_hh    = (const float*)d_in[7];
    const float* b_ih    = (const float*)d_in[8];
    const float* b_hh    = (const float*)d_in[9];
    const float* w_gen   = (const float*)d_in[10];
    const float* b_gen   = (const float*)d_in[11];
    float* out = (float*)d_out;

    char*  ws  = (char*)d_ws;
    size_t off = 0;
    auto carve = [&](size_t bytes) -> void* {
        void* p = ws + off;
        off += (bytes + 255) & ~(size_t)255;
        return p;
    };
    bf16_t* bH16    = (bf16_t*)carve((size_t)BDIM * SDIM * DDIM * 2);   // 67 MB
    bf16_t* bhp16   = (bf16_t*)carve((size_t)BDIM * SDIM * HDIM * 2);   // 67 MB
    bf16_t* wi2h16  = (bf16_t*)carve((size_t)HDIM * DDIM * 2);
    bf16_t* wh2h16  = (bf16_t*)carve((size_t)HDIM * HDIM * 2);
    bf16_t* wcat16  = (bf16_t*)carve((size_t)G4H * KCAT * 2);
    bf16_t* wgen16  = (bf16_t*)carve((size_t)48 * HDIM * 2);
    float*  biascat = (float*) carve((size_t)G4H * 4);
    float*  bgenp   = (float*) carve((size_t)48 * 4);
    bf16_t* hproj16 = (bf16_t*)carve((size_t)BDIM * HDIM * 2);
    float*  alpha   = (float*) carve((size_t)BDIM * SDIM * 4);
    bf16_t* xh      = (bf16_t*)carve((size_t)BDIM * KCAT * 2);
    float*  gates   = (float*) carve((size_t)BDIM * G4H * 4);
    float*  cstate  = (float*) carve((size_t)BDIM * HDIM * 4);
    bf16_t* hidd16  = (bf16_t*)carve((size_t)BDIM * TSTEPS * HDIM * 2); // 13.6 MB
    if (off > ws_size) return;   // ~161 MB required

    const int TB = 256;
    const size_t nBH = (size_t)BDIM * SDIM * DDIM;
    f32_to_bf16<<<(unsigned)((nBH + TB - 1) / TB), TB, 0, stream>>>(batch_H, bH16, nBH);
    f32_to_bf16<<<(HDIM * DDIM + TB - 1) / TB, TB, 0, stream>>>(w_i2h, wi2h16, HDIM * DDIM);
    f32_to_bf16<<<(HDIM * HDIM + TB - 1) / TB, TB, 0, stream>>>(w_h2h, wh2h16, HDIM * HDIM);
    build_wcat<<<(G4H * KCAT + TB - 1) / TB, TB, 0, stream>>>(w_ih, w_hh, wcat16);
    build_small<<<(48 * HDIM + TB - 1) / TB, TB, 0, stream>>>(w_gen, b_gen, b_ih, b_hh,
                                                              wgen16, bgenp, biascat);
    init_state<<<(BDIM * KCAT + TB - 1) / TB, TB, 0, stream>>>(xh, cstate);

    // encoder projection: bhp[B*S, H] = batch_H[B*S, D] @ w_i2h[H, D]^T
    gemm_bf16_wmma<bf16_t><<<dim3(HDIM / BN, (BDIM * SDIM) / BM), TB, 0, stream>>>(
        bH16, DDIM, wi2h16, nullptr, bhp16, HDIM, HDIM, BDIM * SDIM, HDIM, DDIM);

    for (int t = 0; t < TSTEPS; ++t) {
        // hid_proj = h @ w_h2h^T + b_h2h   (h = bf16 slice of xh, lda = KCAT)
        gemm_bf16_wmma<bf16_t><<<dim3(HDIM / BN, BDIM / BM), TB, 0, stream>>>(
            xh + XPAD, KCAT, wh2h16, b_h2h, hproj16, HDIM, HDIM, BDIM, HDIM, HDIM);

        attn_softmax_kernel<<<BDIM, TB, 0, stream>>>(bhp16, hproj16, w_score, alpha);
        context_onehot_kernel<<<BDIM, TB, 0, stream>>>(alpha, bH16, gt, t, xh);

        // gates = [x|h] @ [w_ih|w_hh]^T + (b_ih + b_hh)
        gemm_bf16_wmma<float><<<dim3(G4H / BN, BDIM / BM), TB, 0, stream>>>(
            xh, KCAT, wcat16, biascat, gates, G4H, G4H, BDIM, G4H, KCAT);

        lstm_update_kernel<<<(BDIM * HDIM) / TB, TB, 0, stream>>>(gates, cstate, xh,
                                                                  hidd16, t);
    }

    // probs[B*T, 38] = hiddens[B*T, H] @ w_gen_pad[48, H]^T + b_gen
    gemm_bf16_wmma<float><<<dim3(1, (BDIM * TSTEPS) / BM), TB, 0, stream>>>(
        hidd16, HDIM, wgen16, bgenp, out, CCLS, CCLS, BDIM * TSTEPS, 48, HDIM);
}
